// DifferentialAttention_51092930953782
// MI455X (gfx1250) — matrix operand order repaired and checked by reading, not compiled, not run on hardware
//
#include <hip/hip_runtime.h>
#include <hip/hip_bf16.h>

// ---------------------------------------------------------------------------
// Differential attention, MI455X (gfx1250), wave32 + bf16 WMMA.
//
// Identity: softmax of a permuted row has the same normalizer, so
//   a2 = shift(a1)  and  o = lam * a1 @ (v - roll(v,-1)).
// One softmax, one score matrix, one P@V.
//
// All WMMA operand tiles are laid out so fragment loads are ds_load_b128:
//   - A fragments: row-major M x K tiles (two 16B runs per lane)
//   - B fragments: N-major (N x K) tiles read with the contiguous loader
// Vdiff is stored transposed in global ([B,H,dh,L]) so its LDS tile is N-major.
// Attention K/V tiles use CDNA5 global_load_async_to_lds_b128 (ASYNCcnt).
// ---------------------------------------------------------------------------

#define Bdim 2
#define Ldim 2048
#define Ddim 1024
#define Hdim 16
#define DH   64

typedef __bf16 bf16;
typedef bf16  v16bf __attribute__((ext_vector_type(16)));
typedef float v8f   __attribute__((ext_vector_type(8)));
typedef int   v4i   __attribute__((ext_vector_type(4)));

typedef __attribute__((address_space(1))) v4i* gv4i_p;
typedef __attribute__((address_space(3))) v4i* sv4i_p;

#if defined(__has_builtin)
#  if __has_builtin(__builtin_amdgcn_global_load_async_to_lds_b128)
#    define HAVE_ASYNC_LDS 1
#  endif
#endif
#ifndef HAVE_ASYNC_LDS
#  define HAVE_ASYNC_LDS 0
#endif

static __device__ inline v8f wmma_bf16(v16bf a, v16bf b, v8f c) {
  return __builtin_amdgcn_wmma_f32_16x16x32_bf16(false, a, false, b,
                                                 (short)0, c, false, false);
}

// A fragment (16x32, M x K), row-major LDS source.
// ISA 7.12.2: lanes 0-15 hold K {0..7,16..23}, lanes 16-31 hold K {8..15,24..31}.
static __device__ inline v16bf load_a_frag(const bf16* s, int stride, int lane) {
  v16bf a;
  const int row = lane & 15;
  const int kb  = (lane < 16) ? 0 : 8;
#pragma unroll
  for (int e = 0; e < 16; ++e) {
    int k = kb + ((e >> 3) << 4) + (e & 7);
    a[e] = s[row * stride + k];
  }
  return a;
}

// B fragment (32x16 K x N) from an N-major (N x K row-major) LDS tile:
// lane reads two 8-element bf16 runs -> two ds_load_b128.
// Lanes 0-15: N=lane, K {0..7,16..23}.  Lanes 16-31: N=lane-16, K {8..15,24..31}.
static __device__ inline v16bf load_bT_frag(const bf16* s, int stride, int lane) {
  v16bf b;
  const int n  = lane & 15;
  const int kb = (lane < 16) ? 0 : 8;
#pragma unroll
  for (int e = 0; e < 16; ++e) b[e] = s[n * stride + kb + ((e >> 3) << 4) + (e & 7)];
  return b;
}

// 32B contiguous global->LDS tile copy; async-to-LDS when the toolchain has it.
static __device__ inline void tile_copy16(const bf16* __restrict__ g, bf16* l) {
#if HAVE_ASYNC_LDS
  __builtin_amdgcn_global_load_async_to_lds_b128((gv4i_p)(g), (sv4i_p)(l), 0, 0);
  __builtin_amdgcn_global_load_async_to_lds_b128((gv4i_p)(g + 8), (sv4i_p)(l + 8),
                                                 0, 0);
#else
#pragma unroll
  for (int i = 0; i < 16; ++i) l[i] = g[i];
#endif
}

static __device__ inline void wait_async_tiles() {
#if HAVE_ASYNC_LDS
#  if __has_builtin(__builtin_amdgcn_s_wait_asynccnt)
  __builtin_amdgcn_s_wait_asynccnt(0);
  asm volatile("" ::: "memory");
#  else
  asm volatile("s_wait_asynccnt 0" ::: "memory");
#  endif
#endif
}

// ---------------------------------------------------------------------------
// Kernel 1: qkv = x @ w_qkv^T + b_qkv -> q/k/v bf16 [B,H,L,dh].
// 128 threads (4 waves), tile 64x64, K-step 64 (2 chained WMMAs / acc).
// ---------------------------------------------------------------------------
__global__ __launch_bounds__(128) void qkv_gemm_kernel(
    const float* __restrict__ x, const float* __restrict__ w_qkv,
    const float* __restrict__ b_qkv, bf16* __restrict__ qb,
    bf16* __restrict__ kb_, bf16* __restrict__ vb) {
  __shared__ bf16 As[64][64];   // M x K
  __shared__ bf16 Bs[64][64];   // N x K (direct row copy of w_qkv)
  const int tid = threadIdx.x, wave = tid >> 5, lane = tid & 31;
  const int m0 = blockIdx.x * 64;
  const int n0 = blockIdx.y * 64;
  v8f acc[4] = {};

  for (int k0 = 0; k0 < Ddim; k0 += 64) {
    const int row = tid >> 1, ch = (tid & 1) * 32;
    const float* asrc = x + (size_t)(m0 + row) * Ddim + k0 + ch;
    const float* bsrc = w_qkv + (size_t)(n0 + row) * Ddim + k0 + ch;
#pragma unroll
    for (int i = 0; i < 32; ++i) {
      As[row][ch + i] = (bf16)asrc[i];
      Bs[row][ch + i] = (bf16)bsrc[i];
    }
    __syncthreads();
    v16bf a0 = load_a_frag(&As[wave * 16][0], 64, lane);
    v16bf a1 = load_a_frag(&As[wave * 16][32], 64, lane);
#pragma unroll
    for (int nt = 0; nt < 4; ++nt) {
      acc[nt] = wmma_bf16(a0, load_bT_frag(&Bs[nt * 16][0], 64, lane), acc[nt]);
      acc[nt] = wmma_bf16(a1, load_bT_frag(&Bs[nt * 16][32], 64, lane), acc[nt]);
    }
    __syncthreads();
  }

#pragma unroll
  for (int nt = 0; nt < 4; ++nt) {
    const int ng  = n0 + nt * 16 + (lane & 15);
    const int sel = ng >> 10, d = ng & 1023, h = d >> 6, c = d & 63;
    const float bias = b_qkv[ng];
    bf16* dst = (sel == 0) ? qb : (sel == 1) ? kb_ : vb;
#pragma unroll
    for (int r = 0; r < 8; ++r) {
      const int m = m0 + wave * 16 + ((lane < 16) ? r : 8 + r);
      const int bb = m >> 11, l = m & 2047;
      dst[(((size_t)(bb * Hdim + h)) * Ldim + l) * DH + c] =
          (bf16)(acc[nt][r] + bias);
    }
  }
}

// ---------------------------------------------------------------------------
// Kernel 2: vdiff_t[bh][c][l] = v[bh][l][c] - v[bh][(l+1)%L][c]
// (transpose so the attention V tile is N-major in LDS)
// ---------------------------------------------------------------------------
__global__ __launch_bounds__(256) void vdiff_kernel(const bf16* __restrict__ v,
                                                    bf16* __restrict__ vdt) {
  const size_t idx = (size_t)blockIdx.x * 256 + threadIdx.x;
  const int c = (int)(idx & 63);
  const int l = (int)((idx >> 6) & 2047);
  const size_t bh = idx >> 17;
  const size_t nidx = (bh << 17) | ((size_t)((l + 1) & 2047) << 6) | (size_t)c;
  vdt[(bh << 17) | ((size_t)c << 11) | (size_t)l] =
      (bf16)((float)v[idx] - (float)v[nidx]);
}

// ---------------------------------------------------------------------------
// Kernel 3: flash-style attention.  Block = 64 q-rows x one (b,h).
// ---------------------------------------------------------------------------
__global__ __launch_bounds__(128) void attn_kernel(
    const bf16* __restrict__ qg, const bf16* __restrict__ kg,
    const bf16* __restrict__ vdtg, const float* __restrict__ lam_p,
    bf16* __restrict__ attn) {
  __shared__ bf16 Ks[32][64];    // keys x dh  (N-major for Q@K^T)
  __shared__ bf16 Vst[64][32];   // dh x keys  (N-major for P@Vdiff)
  __shared__ bf16 Ps[4][16][32];
  const int tid = threadIdx.x, wave = tid >> 5, lane = tid & 31;
  const int q0 = blockIdx.x * 64;
  const int bh = blockIdx.y;
  const float scale = 0.03125f;  // d_model^-0.5 = 1/32
  const float lam = lam_p[0];
  const bf16* qb  = qg + ((size_t)bh << 17);
  const bf16* kb  = kg + ((size_t)bh << 17);
  const bf16* vbt = vdtg + ((size_t)bh << 17);

  // Q fragments live in registers: rows q0+wave*16+lane%16, two dh chunks.
  v16bf qa0, qa1;
  {
    const bf16* qr = qb + (size_t)(q0 + wave * 16 + (lane & 15)) * DH;
    const int kbse = (lane < 16) ? 0 : 8;
#pragma unroll
    for (int e = 0; e < 16; ++e) {
      int k = kbse + ((e >> 3) << 4) + (e & 7);
      qa0[e] = qr[k];
      qa1[e] = qr[32 + k];
    }
  }

  // per-thread addresses for cooperative tile staging
  const int krow = tid >> 2, kcs = (tid & 3) * 16;   // K: 32 rows x 64
  const int vrow = tid >> 1, vcs = (tid & 1) * 16;   // V: 64 rows x 32

  float row_max[8], row_sum[8];
#pragma unroll
  for (int r = 0; r < 8; ++r) { row_max[r] = -1e30f; row_sum[r] = 0.f; }
  v8f o_acc[4] = {};

  for (int j0 = 0; j0 < Ldim; j0 += 32) {
    tile_copy16(kb + (size_t)(j0 + krow) * DH + kcs, &Ks[krow][kcs]);
    tile_copy16(vbt + (size_t)vrow * Ldim + j0 + vcs, &Vst[vrow][vcs]);
    if (j0 + 32 < Ldim) {  // prefetch next stream tiles
      __builtin_prefetch(kb + (size_t)(j0 + 32 + krow) * DH + kcs, 0, 0);
      __builtin_prefetch(vbt + (size_t)vrow * Ldim + j0 + 32 + vcs, 0, 0);
    }
    wait_async_tiles();
    __syncthreads();

    // S(16x32) = Q(16x64) @ K^T  (keys are LDS rows -> contiguous B loads)
    v8f s0 = {}, s1 = {};
    s0 = wmma_bf16(qa0, load_bT_frag(&Ks[0][0], 64, lane), s0);
    s0 = wmma_bf16(qa1, load_bT_frag(&Ks[0][32], 64, lane), s0);
    s1 = wmma_bf16(qa0, load_bT_frag(&Ks[16][0], 64, lane), s1);
    s1 = wmma_bf16(qa1, load_bT_frag(&Ks[16][32], 64, lane), s1);

    // Online softmax. C layout: row m<8 -> lanes 0-15 (VGPR m), m>=8 -> 16-31.
#pragma unroll
    for (int r = 0; r < 8; ++r) {
      float a = s0[r] * scale, b = s1[r] * scale;
      float mt = fmaxf(a, b);
#pragma unroll
      for (int off = 8; off > 0; off >>= 1)
        mt = fmaxf(mt, __shfl_xor(mt, off, 32));
      const float nm = fmaxf(row_max[r], mt);
      const float alpha = __expf(row_max[r] - nm);
      row_max[r] = nm;
      const float p0 = __expf(a - nm), p1 = __expf(b - nm);
      s0[r] = p0; s1[r] = p1;
      float rs = p0 + p1;
#pragma unroll
      for (int off = 8; off > 0; off >>= 1) rs += __shfl_xor(rs, off, 32);
      row_sum[r] = row_sum[r] * alpha + rs;
#pragma unroll
      for (int t = 0; t < 4; ++t) o_acc[t][r] *= alpha;
    }

    // C-layout -> A-layout via wave-private LDS (same-wave DS is in-order).
#pragma unroll
    for (int r = 0; r < 8; ++r) {
      const int m = (lane < 16) ? r : 8 + r, n = lane & 15;
      Ps[wave][m][n]      = (bf16)s0[r];
      Ps[wave][m][16 + n] = (bf16)s1[r];
    }
    v16bf pa = load_a_frag(&Ps[wave][0][0], 32, lane);

    // O(16x64) += P(16x32) @ Vdiff(32x64)   (Vst is N-major: contiguous loads)
#pragma unroll
    for (int t = 0; t < 4; ++t)
      o_acc[t] = wmma_bf16(pa, load_bT_frag(&Vst[t * 16][0], 32, lane), o_acc[t]);
    __syncthreads();
  }

  const int h = bh & (Hdim - 1), b = bh >> 4;
#pragma unroll
  for (int r = 0; r < 8; ++r) {
    const int m = q0 + wave * 16 + ((lane < 16) ? r : 8 + r);
    const float inv = lam / row_sum[r];
#pragma unroll
    for (int t = 0; t < 4; ++t) {
      const int c = h * DH + t * 16 + (lane & 15);
      attn[((size_t)b * Ldim + m) * Ddim + c] = (bf16)(o_acc[t][r] * inv);
    }
  }
}

// ---------------------------------------------------------------------------
// Kernel 4: out = attn @ w_out^T + b_out  (fp32 result)
// ---------------------------------------------------------------------------
__global__ __launch_bounds__(128) void out_gemm_kernel(
    const bf16* __restrict__ attn, const float* __restrict__ w_out,
    const float* __restrict__ b_out, float* __restrict__ out) {
  __shared__ bf16 As[64][64];
  __shared__ bf16 Bs[64][64];   // N x K (row copy of w_out)
  const int tid = threadIdx.x, wave = tid >> 5, lane = tid & 31;
  const int m0 = blockIdx.x * 64, n0 = blockIdx.y * 64;
  v8f acc[4] = {};

  for (int k0 = 0; k0 < Ddim; k0 += 64) {
    const int row = tid >> 1, ch = (tid & 1) * 32;
    const bf16* asrc = attn + (size_t)(m0 + row) * Ddim + k0 + ch;
    const float* bsrc = w_out + (size_t)(n0 + row) * Ddim + k0 + ch;
#pragma unroll
    for (int i = 0; i < 32; ++i) {
      As[row][ch + i] = asrc[i];
      Bs[row][ch + i] = (bf16)bsrc[i];
    }
    __syncthreads();
    v16bf a0 = load_a_frag(&As[wave * 16][0], 64, lane);
    v16bf a1 = load_a_frag(&As[wave * 16][32], 64, lane);
#pragma unroll
    for (int nt = 0; nt < 4; ++nt) {
      acc[nt] = wmma_bf16(a0, load_bT_frag(&Bs[nt * 16][0], 64, lane), acc[nt]);
      acc[nt] = wmma_bf16(a1, load_bT_frag(&Bs[nt * 16][32], 64, lane), acc[nt]);
    }
    __syncthreads();
  }

#pragma unroll
  for (int nt = 0; nt < 4; ++nt) {
    const int ng = n0 + nt * 16 + (lane & 15);
    const float bias = b_out[ng];
#pragma unroll
    for (int r = 0; r < 8; ++r) {
      const int m = m0 + wave * 16 + ((lane < 16) ? r : 8 + r);
      out[(size_t)m * Ddim + ng] = acc[nt][r] + bias;
    }
  }
}

// ---------------------------------------------------------------------------
extern "C" void kernel_launch(void* const* d_in, const int* in_sizes, int n_in,
                              void* d_out, int out_size, void* d_ws,
                              size_t ws_size, hipStream_t stream) {
  const float* x     = (const float*)d_in[0];
  const float* w_qkv = (const float*)d_in[1];
  const float* b_qkv = (const float*)d_in[2];
  const float* w_out = (const float*)d_in[3];
  const float* b_out = (const float*)d_in[4];
  const float* lam   = (const float*)d_in[5];

  const size_t NQK = (size_t)Bdim * Hdim * Ldim * DH;  // 4 Mi elems
  bf16* qb   = (bf16*)d_ws;
  bf16* kb   = qb + NQK;
  bf16* vb   = kb + NQK;
  bf16* vdt  = vb + NQK;   // transposed vdiff [B,H,dh,L]
  bf16* attn = vdt + NQK;  // [B,L,D]

  const dim3 blk(128);
  qkv_gemm_kernel<<<dim3(64, 48), blk, 0, stream>>>(x, w_qkv, b_qkv, qb, kb, vb);
  vdiff_kernel<<<dim3((unsigned)(NQK / 256)), dim3(256), 0, stream>>>(vb, vdt);
  attn_kernel<<<dim3(Ldim / 64, Bdim * Hdim), blk, 0, stream>>>(qb, kb, vdt, lam, attn);
  out_gemm_kernel<<<dim3(64, 16), blk, 0, stream>>>(attn, w_out, b_out, (float*)d_out);
}
